// QANetC2QAttention_90477781057878
// MI455X (gfx1250) — compile-verified, matrix operand order checked by reading, not run on hardware
//
#include <hip/hip_runtime.h>
#include <hip/hip_bf16.h>

#define B_ 64
#define C_ 1024
#define Q_ 128
#define H_ 128
#define NEG_INF_ (-1e30f)
#define K3P 56   // padded LDS row stride (halves): 112B, 16B-aligned rows

typedef __attribute__((ext_vector_type(16))) _Float16 v16h;
typedef __attribute__((ext_vector_type(8)))  _Float16 v8h;
typedef __attribute__((ext_vector_type(8)))  float    v8f;

// K index held by (lane, element) for 16-bit A/B fragments of
// v_wmma_*_16x16x32_* per ISA 7.12.2:
//   lanes 0-15 : K = {0..7, 16..23},  lanes 16-31 : K = {8..15, 24..31}
__device__ __forceinline__ int wmma_k(int lane, int e) {
  int k = (e < 8) ? e : (e + 8);
  return k + ((lane & 16) ? 8 : 0);
}

__device__ __forceinline__ v16h cat16(v8h lo, v8h hi) {
  return __builtin_shufflevector(lo, hi, 0, 1, 2, 3, 4, 5, 6, 7,
                                 8, 9, 10, 11, 12, 13, 14, 15);
}

// ---------------------------------------------------------------------------
// k0: rowdot[b,c] = c[b,c,:].w1 ; coldot[b,q] = q[b,q,:].w2 + bias
// ---------------------------------------------------------------------------
__global__ void __launch_bounds__(256) k0_dots(
    const float* __restrict__ cg, const float* __restrict__ qg,
    const float* __restrict__ ww, const float* __restrict__ wb,
    float* __restrict__ rowdot, float* __restrict__ coldot) {
  int wave = threadIdx.x >> 5, lane = threadIdx.x & 31;
  int gid = blockIdx.x * 8 + wave;
  const int NRD = B_ * C_;
  float sum = 0.f;
  if (gid < NRD) {
    const float* row = cg + (size_t)gid * H_;
#pragma unroll
    for (int j = 0; j < 4; ++j) sum += row[lane + 32 * j] * ww[lane + 32 * j];
  } else {
    int idx = gid - NRD;
    const float* row = qg + (size_t)idx * H_;
#pragma unroll
    for (int j = 0; j < 4; ++j) sum += row[lane + 32 * j] * ww[H_ + lane + 32 * j];
  }
#pragma unroll
  for (int off = 16; off >= 1; off >>= 1) sum += __shfl_xor(sum, off, 32);
  if (lane == 0) {
    if (gid < NRD) rowdot[gid] = sum;
    else           coldot[gid - NRD] = sum + wb[0];
  }
}

// k0b: qT16[b,h,q] = (f16) q[b,q,h]  (transposed f16 copy for k4 B-fragments)
__global__ void __launch_bounds__(256) k0b_qT(const float* __restrict__ qg,
                                              _Float16* __restrict__ qT16) {
  size_t i = (size_t)blockIdx.x * 256 + threadIdx.x;   // (b,q,h)
  int h = (int)(i % H_);
  size_t bq = i / H_;
  int qq = (int)(bq % Q_);
  int b  = (int)(bq / Q_);
  qT16[(size_t)(b * H_ + h) * Q_ + qq] = (_Float16)qg[i];
}

// ---------------------------------------------------------------------------
// k1: S[b,c,q] = (c*w3).q^T + rowdot + coldot(+bias); store S fp32.
// Also per-row (axis=q) softmax stats with q_mask: rowmax, rowsum.
// ---------------------------------------------------------------------------
__global__ void __launch_bounds__(256) k1_scores(
    const float* __restrict__ cg, const float* __restrict__ qg,
    const int* __restrict__ qmask, const float* __restrict__ ww,
    const float* __restrict__ rowdot, const float* __restrict__ coldot,
    float* __restrict__ S, float* __restrict__ rowmax, float* __restrict__ rowsum) {
  int b    = blockIdx.y;
  int c0   = blockIdx.x * 16;
  int wave = threadIdx.x >> 5, lane = threadIdx.x & 31;
  int q0   = wave * 16;
  int mrow = lane & 15;
  const float* w3 = ww + 2 * H_;

  v8f acc = {};
#pragma unroll
  for (int kt = 0; kt < 4; ++kt) {
    v16h a, bb;
    const float* arow = cg + (size_t)(b * C_ + c0 + mrow) * H_ + kt * 32;
    const float* brow = qg + (size_t)(b * Q_ + q0 + mrow) * H_ + kt * 32;
#pragma unroll
    for (int e = 0; e < 16; ++e) {
      int k = wmma_k(lane, e);
      a[e]  = (_Float16)(arow[k] * w3[kt * 32 + k]);
      bb[e] = (_Float16)(brow[k]);
    }
    acc = __builtin_amdgcn_wmma_f32_16x16x32_f16(false, a, false, bb,
                                                 (short)0, acc, false, false);
  }

  int n = lane & 15;
  int mbase = (lane & 16) ? 8 : 0;
  float cd = coldot[b * Q_ + q0 + n];
  int   qm = qmask[b * Q_ + q0 + n];
  float sv[8];
#pragma unroll
  for (int i = 0; i < 8; ++i) {
    int m = mbase + i;
    float v = acc[i] + rowdot[b * C_ + c0 + m] + cd;
    S[(size_t)(b * C_ + c0 + m) * Q_ + q0 + n] = v;
    sv[i] = qm ? v : NEG_INF_;
  }

  __shared__ float red[16][8];
  __shared__ float rmax[16];
  float mx[8];
#pragma unroll
  for (int i = 0; i < 8; ++i) {
    float r = sv[i];
#pragma unroll
    for (int off = 8; off >= 1; off >>= 1) r = fmaxf(r, __shfl_xor(r, off, 16));
    mx[i] = r;
  }
  if (lane == 0) {
#pragma unroll
    for (int i = 0; i < 8; ++i) red[i][wave] = mx[i];
  } else if (lane == 16) {
#pragma unroll
    for (int i = 0; i < 8; ++i) red[8 + i][wave] = mx[i];
  }
  __syncthreads();
  if (threadIdx.x < 16) {
    float r = red[threadIdx.x][0];
#pragma unroll
    for (int w = 1; w < 8; ++w) r = fmaxf(r, red[threadIdx.x][w]);
    rmax[threadIdx.x] = r;
  }
  __syncthreads();
  float sm[8];
#pragma unroll
  for (int i = 0; i < 8; ++i) {
    float r = expf(sv[i] - rmax[mbase + i]);
#pragma unroll
    for (int off = 8; off >= 1; off >>= 1) r += __shfl_xor(r, off, 16);
    sm[i] = r;
  }
  if (lane == 0) {
#pragma unroll
    for (int i = 0; i < 8; ++i) red[i][wave] = sm[i];
  } else if (lane == 16) {
#pragma unroll
    for (int i = 0; i < 8; ++i) red[8 + i][wave] = sm[i];
  }
  __syncthreads();
  if (threadIdx.x < 16) {
    float r = 0.f;
#pragma unroll
    for (int w = 0; w < 8; ++w) r += red[threadIdx.x][w];
    rowmax[b * C_ + c0 + threadIdx.x] = rmax[threadIdx.x];
    rowsum[b * C_ + c0 + threadIdx.x] = r;
  }
}

// ---------------------------------------------------------------------------
// k2: column (axis=c) softmax stats with c_mask; online max/sum over C=1024.
// ---------------------------------------------------------------------------
__global__ void __launch_bounds__(128) k2_colstats(
    const float* __restrict__ S, const int* __restrict__ cmask,
    float* __restrict__ colmax, float* __restrict__ colsum) {
  int b = blockIdx.x;
  int q = threadIdx.x;
  __shared__ int cm[C_];
  for (int i = threadIdx.x; i < C_; i += 128) cm[i] = cmask[b * C_ + i];
  __syncthreads();
  float m = -3.0e38f, s = 0.f;
  const float* Sb = S + (size_t)b * C_ * Q_ + q;
  for (int c = 0; c < C_; ++c) {
    __builtin_prefetch(Sb + (size_t)(c + 16) * Q_, 0, 1);  // stream-ahead hint
    float v = cm[c] ? Sb[(size_t)c * Q_] : NEG_INF_;
    float nm = fmaxf(m, v);
    s = s * expf(m - nm) + expf(v - nm);
    m = nm;
  }
  colmax[b * Q_ + q] = m;
  colsum[b * Q_ + q] = s;
}

// ---------------------------------------------------------------------------
// k3: Mpart[slice][b,q,h] = sum_{c in slice} s2[b,c,q] * c[b,c,h]
// LDS tiles staged as f16, TRANSPOSED ([q][c], [h][c], padded stride K3P) so
// each WMMA fragment read is two aligned 16B runs; one convert per element.
// ---------------------------------------------------------------------------
__global__ void __launch_bounds__(256) k3_colctx(
    const float* __restrict__ S, const float* __restrict__ cg,
    const int* __restrict__ cmask, const float* __restrict__ colmax,
    const float* __restrict__ colsum, float* __restrict__ Mpart) {
  int b    = blockIdx.x;
  int ksl  = blockIdx.y;           // 0..3, each covers 256 c-values
  int wave = threadIdx.x >> 5, lane = threadIdx.x & 31;
  int q0   = wave * 16;
  int mrow = lane & 15;
  int koff = (lane & 16) ? 8 : 0;

  __shared__ __align__(16) _Float16 s2T[Q_][K3P];  // [q][c-in-tile]
  __shared__ __align__(16) _Float16 cT[H_][K3P];   // [h][c-in-tile]
  __shared__ float cmx[Q_], rcs[Q_];
  for (int i = threadIdx.x; i < Q_; i += 256) {
    cmx[i] = colmax[b * Q_ + i];
    rcs[i] = 1.f / colsum[b * Q_ + i];
  }
  __syncthreads();

  v8f acc[8] = {};
  for (int kt = ksl * 8; kt < ksl * 8 + 8; ++kt) {
    __syncthreads();
    int kc0 = kt * 32;
#pragma unroll
    for (int r = 0; r < 16; ++r) {
      int idx = r * 256 + threadIdx.x;
      int cc = idx >> 7, qq = idx & 127;
      int gc = kc0 + cc;
      float sval = S[(size_t)(b * C_ + gc) * Q_ + qq];
      float masked = cmask[b * C_ + gc] ? sval : NEG_INF_;
      s2T[qq][cc] = (_Float16)(expf(masked - cmx[qq]) * rcs[qq]);
      cT[qq][cc]  = (_Float16)cg[(size_t)(b * C_ + gc) * H_ + qq];
    }
    __syncthreads();
    v8h alo = *(const v8h*)&s2T[q0 + mrow][koff];
    v8h ahi = *(const v8h*)&s2T[q0 + mrow][koff + 16];
    v16h a = cat16(alo, ahi);
#pragma unroll
    for (int ht = 0; ht < 8; ++ht) {
      v8h blo = *(const v8h*)&cT[ht * 16 + mrow][koff];
      v8h bhi = *(const v8h*)&cT[ht * 16 + mrow][koff + 16];
      v16h bb = cat16(blo, bhi);
      acc[ht] = __builtin_amdgcn_wmma_f32_16x16x32_f16(false, a, false, bb,
                                                       (short)0, acc[ht], false, false);
    }
  }

  float* Mo = Mpart + (size_t)ksl * B_ * Q_ * H_;
  int n = lane & 15;
  int mbase = (lane & 16) ? 8 : 0;
#pragma unroll
  for (int ht = 0; ht < 8; ++ht) {
#pragma unroll
    for (int i = 0; i < 8; ++i)
      Mo[(size_t)(b * Q_ + q0 + mbase + i) * H_ + ht * 16 + n] = acc[ht][i];
  }
}

// k3r: M reduction, emitted directly as f16 TRANSPOSED (MT16[b,h,q]) for k4.
__global__ void __launch_bounds__(256) k3r_reduce(const float* __restrict__ Mpart,
                                                  _Float16* __restrict__ MT16) {
  size_t i = (size_t)blockIdx.x * 256 + threadIdx.x;   // (b,q,h)
  const size_t nn = (size_t)B_ * Q_ * H_;
  float v = Mpart[i] + Mpart[i + nn] + Mpart[i + 2 * nn] + Mpart[i + 3 * nn];
  int h = (int)(i % H_);
  size_t bq = i / H_;
  int qq = (int)(bq % Q_);
  int b  = (int)(bq / Q_);
  MT16[(size_t)(b * H_ + h) * Q_ + qq] = (_Float16)v;
}

// ---------------------------------------------------------------------------
// k4: s1 tile (16 x 128) as f16 in LDS (one convert, shared by all 8 waves);
// B-fragments from f16 transposed global copies (contiguous, L2-resident).
// a = s1@q, bmat = s1@M; write concat([c, a, c*a, c*bmat]).
// ---------------------------------------------------------------------------
__global__ void __launch_bounds__(256) k4_output(
    const float* __restrict__ cg,
    const _Float16* __restrict__ qT16, const _Float16* __restrict__ MT16,
    const int* __restrict__ qmask,
    const float* __restrict__ S, const float* __restrict__ rowmax,
    const float* __restrict__ rowsum, float* __restrict__ out) {
  int b    = blockIdx.y;
  int c0   = blockIdx.x * 16;
  int wave = threadIdx.x >> 5, lane = threadIdx.x & 31;

  __shared__ __align__(16) _Float16 s1h[16][Q_];
#pragma unroll
  for (int r = 0; r < 8; ++r) {
    int idx = r * 256 + threadIdx.x;
    int rr = idx >> 7, qq = idx & 127;
    float sval = S[(size_t)(b * C_ + c0 + rr) * Q_ + qq];
    float masked = qmask[b * Q_ + qq] ? sval : NEG_INF_;
    s1h[rr][qq] =
        (_Float16)(expf(masked - rowmax[b * C_ + c0 + rr]) / rowsum[b * C_ + c0 + rr]);
  }
  __syncthreads();

  int h0 = wave * 16;
  int mrow = lane & 15;
  int koff = (lane & 16) ? 8 : 0;
  const _Float16* qrow = qT16 + (size_t)(b * H_ + h0 + mrow) * Q_;
  const _Float16* mrp  = MT16 + (size_t)(b * H_ + h0 + mrow) * Q_;

  v8f acca = {}, accb = {};
#pragma unroll
  for (int kt = 0; kt < 4; ++kt) {
    int kbase = kt * 32 + koff;
    v8h alo = *(const v8h*)&s1h[mrow][kbase];
    v8h ahi = *(const v8h*)&s1h[mrow][kbase + 16];
    v16h a = cat16(alo, ahi);
    v8h qlo = *(const v8h*)(qrow + kbase);
    v8h qhi = *(const v8h*)(qrow + kbase + 16);
    v16h fq = cat16(qlo, qhi);
    v8h mlo = *(const v8h*)(mrp + kbase);
    v8h mhi = *(const v8h*)(mrp + kbase + 16);
    v16h fm = cat16(mlo, mhi);
    acca = __builtin_amdgcn_wmma_f32_16x16x32_f16(false, a, false, fq,
                                                  (short)0, acca, false, false);
    accb = __builtin_amdgcn_wmma_f32_16x16x32_f16(false, a, false, fm,
                                                  (short)0, accb, false, false);
  }

  int n = lane & 15;
  int mbase = (lane & 16) ? 8 : 0;
#pragma unroll
  for (int i = 0; i < 8; ++i) {
    int row = c0 + mbase + i;
    int h = h0 + n;
    float cv = cg[(size_t)(b * C_ + row) * H_ + h];
    size_t ob = (size_t)(b * C_ + row) * (4 * H_);
    float av = acca[i], bv = accb[i];
    out[ob + h]          = cv;
    out[ob + H_ + h]     = av;
    out[ob + 2 * H_ + h] = cv * av;
    out[ob + 3 * H_ + h] = cv * bv;
  }
}

// ---------------------------------------------------------------------------
extern "C" void kernel_launch(void* const* d_in, const int* in_sizes, int n_in,
                              void* d_out, int out_size, void* d_ws, size_t ws_size,
                              hipStream_t stream) {
  (void)in_sizes; (void)n_in; (void)out_size; (void)ws_size;
  const float* cg    = (const float*)d_in[0];
  const float* qg    = (const float*)d_in[1];
  const int*   cmask = (const int*)d_in[2];
  const int*   qmask = (const int*)d_in[3];
  const float* ww    = (const float*)d_in[4];
  const float* wb    = (const float*)d_in[5];
  float* out = (float*)d_out;
  float* ws  = (float*)d_ws;

  // workspace layout: ~54 MB total
  float* S      = ws;                                   // B*C*Q        f32
  float* rowdot = S      + (size_t)B_ * C_ * Q_;        // B*C          f32
  float* coldot = rowdot + (size_t)B_ * C_;             // B*Q          f32
  float* rowmax = coldot + (size_t)B_ * Q_;             // B*C          f32
  float* rowsum = rowmax + (size_t)B_ * C_;             // B*C          f32
  float* colmax = rowsum + (size_t)B_ * C_;             // B*Q          f32
  float* colsum = colmax + (size_t)B_ * Q_;             // B*Q          f32
  float* Mpart  = colsum + (size_t)B_ * Q_;             // 4*B*Q*H      f32
  _Float16* qT16 = (_Float16*)(Mpart + (size_t)4 * B_ * Q_ * H_);  // B*H*Q f16
  _Float16* MT16 = qT16 + (size_t)B_ * H_ * Q_;                    // B*H*Q f16

  k0_dots   <<<(B_ * C_ + B_ * Q_) / 8, 256, 0, stream>>>(cg, qg, ww, wb, rowdot, coldot);
  k0b_qT    <<<(B_ * Q_ * H_) / 256, 256, 0, stream>>>(qg, qT16);
  k1_scores <<<dim3(C_ / 16, B_), 256, 0, stream>>>(cg, qg, qmask, ww, rowdot, coldot,
                                                    S, rowmax, rowsum);
  k2_colstats<<<B_, 128, 0, stream>>>(S, cmask, colmax, colsum);
  k3_colctx <<<dim3(B_, 4), 256, 0, stream>>>(S, cg, cmask, colmax, colsum, Mpart);
  k3r_reduce<<<(B_ * Q_ * H_) / 256, 256, 0, stream>>>(Mpart, MT16);
  k4_output <<<dim3(C_ / 16, B_), 256, 0, stream>>>(cg, qT16, MT16, qmask,
                                                    S, rowmax, rowsum, out);
}
